// PGENet_88244398063759
// MI455X (gfx1250) — matrix operand order repaired and checked by reading, not compiled, not run on hardware
//
#include <hip/hip_runtime.h>
#include <hip/hip_bf16.h>
#include <math.h>
#include <stdint.h>

// ---------------------------------------------------------------------------
// Problem constants (from reference)
// ---------------------------------------------------------------------------
#define BB 8
#define CC 128
#define HH 128
#define WW 128
#define HWW 16384         // H*W
#define EE 4
#define RR 64
#define ER 256            // E*R

// Padded LDS row strides (in halves): +8 halves (16B) kills ds bank conflicts
#define LDC  (CC + 8)     // 136 for 128-wide K rows
#define LDER (ER + 8)     // 264 for 256-wide K rows

// Async global->LDS staging (CDNA5 GLOBAL_LOAD_ASYNC_TO_LDS_B128, ASYNCcnt).
#ifndef USE_ASYNC_LDS
#define USE_ASYNC_LDS 1
#endif

typedef _Float16 half_t;
typedef __attribute__((ext_vector_type(16))) _Float16 v16h;
typedef __attribute__((ext_vector_type(8)))  _Float16 v8h;
typedef __attribute__((ext_vector_type(8)))  float    v8f;

// ---------------------------------------------------------------------------
// WMMA fragment helpers (CDNA5 wave32, V_WMMA_F32_16X16X32_F16 layouts)
// ---------------------------------------------------------------------------
__device__ __forceinline__ v16h load_a_frag(const half_t* __restrict__ base, int ld,
                                            int m0, int k0, int lane) {
    int row  = m0 + (lane & 15);
    int koff = k0 + ((lane >> 4) ? 8 : 0);
    const half_t* p = base + (size_t)row * ld + koff;
    v8h lo = *(const v8h*)(p);
    v8h hi = *(const v8h*)(p + 16);
    v16h r;
#pragma unroll
    for (int i = 0; i < 8; ++i) { r[i] = lo[i]; r[8 + i] = hi[i]; }
    return r;
}

// B fragment out of an LDS tile laid out [ncols][ld] (K contiguous per column)
__device__ __forceinline__ v16h lds_b_frag(const half_t* tile, int ld,
                                           int nloc, int k0, int lane) {
    int col  = nloc + (lane & 15);
    int koff = k0 + ((lane >> 4) ? 8 : 0);
    const half_t* p = tile + col * ld + koff;
    v8h lo = *(const v8h*)(p);
    v8h hi = *(const v8h*)(p + 16);
    v16h r;
#pragma unroll
    for (int i = 0; i < 8; ++i) { r[i] = lo[i]; r[8 + i] = hi[i]; }
    return r;
}

__device__ __forceinline__ v8f wmma_f16(v16h a, v16h b, v8f c) {
    return __builtin_amdgcn_wmma_f32_16x16x32_f16(false, a, false, b, (short)0, c,
                                                  false, false);
}

// ---------------------------------------------------------------------------
// Async 16B global -> LDS copy (one lane = one b128 chunk)
// ---------------------------------------------------------------------------
#if USE_ASYNC_LDS
__device__ __forceinline__ void async_copy16(half_t* lds_dst, const half_t* gsrc) {
    uint32_t lds_addr = (uint32_t)(uintptr_t)lds_dst;   // low 32 bits = LDS offset
    asm volatile("global_load_async_to_lds_b128 %0, %1, off"
                 :: "v"(lds_addr), "v"(gsrc) : "memory");
}
__device__ __forceinline__ void async_wait() {
    asm volatile("s_wait_asynccnt 0x0" ::: "memory");
}
#endif

// Stage ROWS rows of ROWH halves (contiguous in global) into padded LDS tile.
template <int ROWS, int ROWH, int LD>
__device__ __forceinline__ void stage_tile(half_t* lds, const half_t* gsrc, int tid) {
    constexpr int CPR     = ROWH / 8;        // 16B chunks per row
    constexpr int NCHUNKS = ROWS * CPR;
#pragma unroll 2
    for (int i = tid; i < NCHUNKS; i += 256) {
        const int row = i / CPR;
        const int cin = i % CPR;
        half_t* dst = lds + row * LD + cin * 8;
#if USE_ASYNC_LDS
        async_copy16(dst, gsrc + i * 8);
#else
        *(v8h*)dst = *(const v8h*)(gsrc + i * 8);
#endif
    }
}

__device__ __forceinline__ void stage_fence() {
#if USE_ASYNC_LDS
    async_wait();
#endif
    __syncthreads();
}

__device__ __forceinline__ float gelu_exact(float v) {
    return 0.5f * v * (1.0f + erff(v * 0.7071067811865475f));
}

// ---------------------------------------------------------------------------
// Kernel 1: per-(b,c) plane reductions: pooled = mean(x), fe0 = mean(gelu(hp3x3(x)))
// ---------------------------------------------------------------------------
__global__ void pool_fe_kernel(const float* __restrict__ x,
                               float* __restrict__ pooled,
                               float* __restrict__ fe0) {
    __shared__ float s_x[256];
    __shared__ float s_h[256];
    const int plane = blockIdx.x;                  // b*C + c
    const float* p = x + (size_t)plane * HWW;
    const int tid = threadIdx.x;

    float sx = 0.f, sh = 0.f;
    for (int idx = tid; idx < HWW; idx += 256) {
        const int i = idx >> 7;
        const int j = idx & 127;
        const float c = p[idx];
        float nsum = 0.f;
#pragma unroll
        for (int di = -1; di <= 1; ++di) {
#pragma unroll
        for (int dj = -1; dj <= 1; ++dj) {
            if (di == 0 && dj == 0) continue;
            const int ii = i + di, jj = j + dj;
            if (ii >= 0 && ii < HH && jj >= 0 && jj < WW)
                nsum += p[ii * WW + jj];
        }
        }
        const float hp = 8.f * c - nsum;
        sx += c;
        sh += gelu_exact(hp);
    }
    s_x[tid] = sx; s_h[tid] = sh;
    __syncthreads();
    for (int s = 128; s > 0; s >>= 1) {
        if (tid < s) { s_x[tid] += s_x[tid + s]; s_h[tid] += s_h[tid + s]; }
        __syncthreads();
    }
    if (tid == 0) {
        pooled[plane] = s_x[0] * (1.f / HWW);
        fe0[plane]    = s_h[0] * (1.f / HWW);
    }
}

// ---------------------------------------------------------------------------
// Kernel 2: per-b routing: MLP(fe0) -> logits -> softmax -> top-2 gates, gsum
// ---------------------------------------------------------------------------
__global__ void route_kernel(const float* __restrict__ pooled,
                             const float* __restrict__ fe0,
                             const float* __restrict__ mlp_w1,   // [C,2C]
                             const float* __restrict__ mlp_b1,   // [2C]
                             const float* __restrict__ mlp_w2,   // [2C,C]
                             const float* __restrict__ mlp_b2,   // [C]
                             const float* __restrict__ gate_w,   // [C,E]
                             const float* __restrict__ fgate_w,  // [C,E]
                             float* __restrict__ gates,          // [B,E]
                             float* __restrict__ gsum) {         // [B]
    __shared__ float s_fe0[CC];
    __shared__ float s_h1[2 * CC];
    __shared__ float s_fe[CC];
    __shared__ float s_logit[EE];
    const int b = blockIdx.x;
    const int tid = threadIdx.x;

    if (tid < CC) s_fe0[tid] = fe0[b * CC + tid];
    __syncthreads();

    {   // h1 = gelu(fe0 @ w1 + b1), 256 outputs
        float acc = mlp_b1[tid];
        for (int c = 0; c < CC; ++c) acc += s_fe0[c] * mlp_w1[c * (2 * CC) + tid];
        s_h1[tid] = gelu_exact(acc);
    }
    __syncthreads();

    if (tid < CC) {  // fe = h1 @ w2 + b2
        float acc = mlp_b2[tid];
        for (int j = 0; j < 2 * CC; ++j) acc += s_h1[j] * mlp_w2[j * CC + tid];
        s_fe[tid] = acc;
    }
    __syncthreads();

    if (tid < EE) {
        float acc = 0.f;
        for (int c = 0; c < CC; ++c) {
            acc += pooled[b * CC + c] * gate_w[c * EE + tid];
            acc += s_fe[c] * fgate_w[c * EE + tid];
        }
        s_logit[tid] = acc;
    }
    __syncthreads();

    if (tid == 0) {
        float mx = s_logit[0];
        for (int e = 1; e < EE; ++e) mx = fmaxf(mx, s_logit[e]);
        float sc[EE], den = 0.f;
        for (int e = 0; e < EE; ++e) { sc[e] = __expf(s_logit[e] - mx); den += sc[e]; }
        for (int e = 0; e < EE; ++e) sc[e] /= den;
        int i0 = 0;
        for (int e = 1; e < EE; ++e) if (sc[e] > sc[i0]) i0 = e;
        int i1 = -1;
        for (int e = 0; e < EE; ++e) {
            if (e == i0) continue;
            if (i1 < 0 || sc[e] > sc[i1]) i1 = e;
        }
        float gs = 0.f;
        for (int e = 0; e < EE; ++e) {
            float g = (e == i0 || e == i1) ? sc[e] : 0.f;
            gates[b * EE + e] = g;
            gs += g;
        }
        gsum[b] = gs;
    }
}

// ---------------------------------------------------------------------------
// Kernel 3: generic fp32 -> f16 convert
// ---------------------------------------------------------------------------
__global__ void cvt_f16_kernel(const float* __restrict__ in, half_t* __restrict__ out, int n) {
    int i = blockIdx.x * blockDim.x + threadIdx.x;
    if (i < n) out[i] = (half_t)in[i];
}

// ---------------------------------------------------------------------------
// Kernel 4: PW[d,(e,r)] = sum_c proj_w[d,c] * exp_w2[e,c,r]   (batch independent)
// ---------------------------------------------------------------------------
__global__ void pw_fold_kernel(const float* __restrict__ proj_w,   // [C,C]
                               const float* __restrict__ exp_w2,   // [E,C,R]
                               half_t* __restrict__ PWh) {         // [C,ER]
    const int t = blockIdx.x * blockDim.x + threadIdx.x;   // [0, 128*256)
    const int d  = t >> 8;
    const int er = t & 255;
    const int e  = er >> 6;
    const int r  = er & 63;
    float acc = 0.f;
    for (int c = 0; c < CC; ++c)
        acc += proj_w[d * CC + c] * exp_w2[e * (CC * RR) + c * RR + r];
    PWh[d * ER + er] = (half_t)acc;
}

// ---------------------------------------------------------------------------
// Kernel 5: LDS-tiled transpose + f16 convert: in[b][c][hw] -> out[b][hw][c]
// ---------------------------------------------------------------------------
__global__ void transpose_f16_kernel(const float* __restrict__ in, half_t* __restrict__ out) {
    __shared__ float tile[CC * 65];
    const int b   = blockIdx.z;
    const int hw0 = blockIdx.x * 64;
    const int tid = threadIdx.x;
    const float* ib = in + (size_t)b * CC * HWW;
    half_t* ob = out + (size_t)b * HWW * CC;

    for (int idx = tid; idx < CC * 64; idx += 256) {      // coalesced reads along hw
        const int c  = idx >> 6;
        const int hw = idx & 63;
        tile[c * 65 + hw] = ib[(size_t)c * HWW + hw0 + hw];
    }
    __syncthreads();
    for (int idx = tid; idx < CC * 64; idx += 256) {      // coalesced writes along c
        const int hw = idx >> 7;
        const int c  = idx & 127;
        ob[(size_t)(hw0 + hw) * CC + c] = (half_t)tile[c * 65 + hw];
    }
}

// ---------------------------------------------------------------------------
// Kernel 6: expert up-projection + gated product:
//   AG[b][hw][er] = gates[b,e] * (W0 @ X) * silu(W1 @ S)
// grid = (HW/32, 1, B), 256 threads (8 waves); wave -> 32 er-rows x 32 cols
// (2 M-tiles x 2 N-tiles: 8 accs, each LDS B-fragment feeds 2 WMMAs).
// ---------------------------------------------------------------------------
__global__ void expert_up_kernel(const half_t* __restrict__ W0h,  // [ER,C]
                                 const half_t* __restrict__ W1h,  // [ER,C]
                                 const half_t* __restrict__ Xt,   // [B,HW,C]
                                 const half_t* __restrict__ St,   // [B,HW,C]
                                 const float*  __restrict__ gates,
                                 half_t* __restrict__ AG) {       // [B,HW,ER]
    __shared__ half_t xs[32 * LDC];
    __shared__ half_t ss[32 * LDC];
    const int b    = blockIdx.z;
    const int n0   = blockIdx.x * 32;
    const int tid  = threadIdx.x;
    const int wave = tid >> 5;
    const int lane = tid & 31;
    const int m0   = wave * 32;                           // er rows: 8 waves x 32 = 256

    stage_tile<32, CC, LDC>(xs, Xt + ((size_t)b * HWW + n0) * CC, tid);
    stage_tile<32, CC, LDC>(ss, St + ((size_t)b * HWW + n0) * CC, tid);
    stage_fence();

    v8f aa[2][2] = {{{0}}}, gg[2][2] = {{{0}}};
#pragma unroll
    for (int k0 = 0; k0 < CC; k0 += 32) {
        v16h a0[2], a1[2];
#pragma unroll
        for (int mt = 0; mt < 2; ++mt) {
            a0[mt] = load_a_frag(W0h, CC, m0 + 16 * mt, k0, lane);
            a1[mt] = load_a_frag(W1h, CC, m0 + 16 * mt, k0, lane);
        }
#pragma unroll
        for (int j = 0; j < 2; ++j) {
            v16h bx = lds_b_frag(xs, LDC, 16 * j, k0, lane);
            v16h bs = lds_b_frag(ss, LDC, 16 * j, k0, lane);
#pragma unroll
            for (int mt = 0; mt < 2; ++mt) {
                aa[mt][j] = wmma_f16(a0[mt], bx, aa[mt][j]);
                gg[mt][j] = wmma_f16(a1[mt], bs, gg[mt][j]);
            }
        }
    }

    const float gate = gates[b * EE + (m0 >> 6)];         // 32-row chunk: one expert
    half_t* AGb = AG + (size_t)b * HWW * ER;
#pragma unroll
    for (int mt = 0; mt < 2; ++mt) {
        const int rbase = m0 + 16 * mt + ((lane >> 4) ? 8 : 0);
#pragma unroll
        for (int j = 0; j < 2; ++j) {
            const int col = n0 + 16 * j + (lane & 15);
            v8h outp;
#pragma unroll
            for (int i = 0; i < 8; ++i) {
                const float a = aa[mt][j][i];
                const float g = gg[mt][j][i];
                const float s = g / (1.f + __expf(-g));   // silu
                outp[i] = (half_t)(gate * a * s);
            }
            *(v8h*)(AGb + (size_t)col * ER + rbase) = outp;  // contiguous b128 store
        }
    }
}

// ---------------------------------------------------------------------------
// Kernel 7: fused down-projection + proj_out + gated residual:
//   OUT[b][d][hw] = PW[128,256] @ AG[b] + gsum[b] * (Proj[128,128] @ X[b])
// grid = (HW/64, 1, B), 256 threads; two 4-wave column groups; each wave
// 32 d-rows x 32 cols (2 M-tiles x 2 N-tiles -> 2x B-fragment reuse).
// ---------------------------------------------------------------------------
__global__ void down_proj_kernel(const half_t* __restrict__ PWh,    // [C,ER]
                                 const half_t* __restrict__ Projh,  // [C,C]
                                 const half_t* __restrict__ AG,     // [B,HW,ER]
                                 const half_t* __restrict__ Xt,     // [B,HW,C]
                                 const float*  __restrict__ gsum,
                                 float* __restrict__ out) {         // [B,C,HW]
    __shared__ half_t ags[64 * LDER];   // 33.8 KB
    __shared__ half_t xs[64 * LDC];     // 17.4 KB
    const int b    = blockIdx.z;
    const int n0   = blockIdx.x * 64;
    const int tid  = threadIdx.x;
    const int wave = tid >> 5;
    const int lane = tid & 31;
    const int ng   = (wave >> 2) * 32;                     // column group offset (0/32)
    const int m0   = (wave & 3) * 32;                      // d rows: 4 waves x 32 = 128

    stage_tile<64, ER, LDER>(ags, AG + ((size_t)b * HWW + n0) * ER, tid);
    stage_tile<64, CC, LDC>(xs, Xt + ((size_t)b * HWW + n0) * CC, tid);
    stage_fence();

    v8f acc1[2][2] = {{{0}}}, acc2[2][2] = {{{0}}};
#pragma unroll 4
    for (int k0 = 0; k0 < ER; k0 += 32) {
        v16h a[2];
        a[0] = load_a_frag(PWh, ER, m0, k0, lane);
        a[1] = load_a_frag(PWh, ER, m0 + 16, k0, lane);
#pragma unroll
        for (int j = 0; j < 2; ++j) {
            v16h bb = lds_b_frag(ags, LDER, ng + 16 * j, k0, lane);
#pragma unroll
            for (int mt = 0; mt < 2; ++mt)
                acc1[mt][j] = wmma_f16(a[mt], bb, acc1[mt][j]);
        }
    }
#pragma unroll
    for (int k0 = 0; k0 < CC; k0 += 32) {
        v16h a[2];
        a[0] = load_a_frag(Projh, CC, m0, k0, lane);
        a[1] = load_a_frag(Projh, CC, m0 + 16, k0, lane);
#pragma unroll
        for (int j = 0; j < 2; ++j) {
            v16h bb = lds_b_frag(xs, LDC, ng + 16 * j, k0, lane);
#pragma unroll
            for (int mt = 0; mt < 2; ++mt)
                acc2[mt][j] = wmma_f16(a[mt], bb, acc2[mt][j]);
        }
    }

    const float gs = gsum[b];
    float* ob = out + (size_t)b * CC * HWW;
#pragma unroll
    for (int mt = 0; mt < 2; ++mt) {
        const int rbase = m0 + 16 * mt + ((lane >> 4) ? 8 : 0);
#pragma unroll
        for (int j = 0; j < 2; ++j) {
            const int col = n0 + ng + 16 * j + (lane & 15);
#pragma unroll
            for (int i = 0; i < 8; ++i)
                ob[(size_t)(rbase + i) * HWW + col] = acc1[mt][j][i] + gs * acc2[mt][j][i];
        }
    }
}

// ---------------------------------------------------------------------------
// Launch
// ---------------------------------------------------------------------------
extern "C" void kernel_launch(void* const* d_in, const int* in_sizes, int n_in,
                              void* d_out, int out_size, void* d_ws, size_t ws_size,
                              hipStream_t stream) {
    const float* x       = (const float*)d_in[0];
    const float* shared  = (const float*)d_in[1];
    const float* mlp_w1  = (const float*)d_in[2];
    const float* mlp_b1  = (const float*)d_in[3];
    const float* mlp_w2  = (const float*)d_in[4];
    const float* mlp_b2  = (const float*)d_in[5];
    const float* gate_w  = (const float*)d_in[6];
    const float* fgate_w = (const float*)d_in[7];
    const float* exp_w0  = (const float*)d_in[8];
    const float* exp_w1  = (const float*)d_in[9];
    const float* exp_w2  = (const float*)d_in[10];
    const float* proj_w  = (const float*)d_in[11];
    float* out = (float*)d_out;

    // Workspace carve-up (aligned to 256B)
    char* ws = (char*)d_ws;
    size_t ofs = 0;
    auto alloc = [&](size_t bytes) {
        void* p = ws + ofs;
        ofs += (bytes + 255) & ~(size_t)255;
        return p;
    };
    half_t* Xt     = (half_t*)alloc((size_t)BB * HWW * CC * sizeof(half_t)); // 32 MB
    half_t* St     = (half_t*)alloc((size_t)BB * HWW * CC * sizeof(half_t)); // 32 MB
    half_t* AG     = (half_t*)alloc((size_t)BB * HWW * ER * sizeof(half_t)); // 64 MB
    half_t* W0h    = (half_t*)alloc((size_t)ER * CC * sizeof(half_t));
    half_t* W1h    = (half_t*)alloc((size_t)ER * CC * sizeof(half_t));
    half_t* PWh    = (half_t*)alloc((size_t)CC * ER * sizeof(half_t));
    half_t* Projh  = (half_t*)alloc((size_t)CC * CC * sizeof(half_t));
    float*  pooled = (float*)alloc((size_t)BB * CC * sizeof(float));
    float*  fe0    = (float*)alloc((size_t)BB * CC * sizeof(float));
    float*  gates  = (float*)alloc((size_t)BB * EE * sizeof(float));
    float*  gsum   = (float*)alloc((size_t)BB * sizeof(float));
    (void)ws_size; (void)n_in; (void)in_sizes; (void)out_size;

    // 1) per-plane pooling + high-pass GELU mean
    pool_fe_kernel<<<dim3(BB * CC), 256, 0, stream>>>(x, pooled, fe0);

    // 2) routing (MLP + softmax + top-2)
    route_kernel<<<dim3(BB), 256, 0, stream>>>(pooled, fe0, mlp_w1, mlp_b1, mlp_w2,
                                               mlp_b2, gate_w, fgate_w, gates, gsum);

    // 3) weight conversions to f16
    cvt_f16_kernel<<<(ER * CC + 255) / 256, 256, 0, stream>>>(exp_w0, W0h, ER * CC);
    cvt_f16_kernel<<<(ER * CC + 255) / 256, 256, 0, stream>>>(exp_w1, W1h, ER * CC);
    cvt_f16_kernel<<<(CC * CC + 255) / 256, 256, 0, stream>>>(proj_w, Projh, CC * CC);

    // 4) fold proj into exp_w2 (batch-independent)
    pw_fold_kernel<<<dim3(CC), 256, 0, stream>>>(proj_w, exp_w2, PWh);

    // 5) transpose + convert activations to [b][hw][c] f16
    transpose_f16_kernel<<<dim3(HWW / 64, 1, BB), 256, 0, stream>>>(x, Xt);
    transpose_f16_kernel<<<dim3(HWW / 64, 1, BB), 256, 0, stream>>>(shared, St);

    // 6) expert up-projection + gated product  (WMMA + async LDS staging)
    expert_up_kernel<<<dim3(HWW / 32, 1, BB), 256, 0, stream>>>(W0h, W1h, Xt, St,
                                                                gates, AG);

    // 7) fused down-projection + proj_out + gated residual (WMMA + async LDS staging)
    down_proj_kernel<<<dim3(HWW / 64, 1, BB), 256, 0, stream>>>(PWh, Projh, AG, Xt,
                                                                 gsum, out);
}
